// LSTM_24378234372579
// MI455X (gfx1250) — matrix operand order repaired and checked by reading, not compiled, not run on hardware
//
#include <hip/hip_runtime.h>

typedef __attribute__((ext_vector_type(16))) _Float16 v16h;
typedef __attribute__((ext_vector_type(8)))  float    v8f;

namespace {
constexpr int kB = 2048;
constexpr int kT = 2048;
constexpr int kH = 16;
constexpr int kO = 26;
constexpr int kBatchPerBlock = 128;   // 8 waves * 16 batch per wave
}

__device__ __forceinline__ float tanh_fast(float x) {
#if __has_builtin(__builtin_amdgcn_tanhf)
  return __builtin_amdgcn_tanhf(x);            // gfx1250 v_tanh_f32
#else
  float e = __builtin_amdgcn_exp2f(-2.885390081777927f * x);  // exp(-2x)
  return (1.0f - e) * __builtin_amdgcn_rcpf(1.0f + e);
#endif
}

__device__ __forceinline__ float sigmoid_fast(float x) {
  // sigmoid(x) = 0.5*tanh(x/2) + 0.5  -> single transcendental
  return __builtin_fmaf(0.5f, tanh_fast(0.5f * x), 0.5f);
}

// One wave owns a 16-batch tile. Per timestep:
//   g^T[64x16] = W_hh[64x16] * h^T[16x16]  via 4x v_wmma_f32_16x16x32_f16,
//   C-accumulator = x*w_ih + (b_ih+b_hh) in f32,
// then elementwise gates + state update entirely in registers.
__launch_bounds__(256, 1)
__global__ void lstm_recurrence_kernel(const float* __restrict__ x,
                                       const float* __restrict__ w_ih,
                                       const float* __restrict__ w_hh,
                                       const float* __restrict__ b_ih,
                                       const float* __restrict__ b_hh,
                                       const float* __restrict__ w_out,
                                       const float* __restrict__ b_out,
                                       float* __restrict__ out) {
  __shared__ float h_sm[kBatchPerBlock][kH];

  const int tid       = threadIdx.x;
  const int lane      = tid & 31;
  const int waveInBlk = tid >> 5;
  const int half      = lane >> 4;   // 0: lanes 0-15, 1: lanes 16-31
  const int m         = lane & 15;   // batch index within tile / gate row (A)
  const int batch     = blockIdx.x * kBatchPerBlock + waveInBlk * 16 + m;

  // ---- Preload A tiles (w_hh rows as f16). Hidden j maps to K=j (K 0..15).
  // In the 16-bit A layout that means: lane half h holds w_hh[16t+m][8h+e]
  // in elements e=0..7; elements 8..15 (K 16..31) are zero.
  v16h A[4];
#pragma unroll
  for (int t = 0; t < 4; ++t) {
    v16h a = {};
    {
      const float* wr = w_hh + (16 * t + m) * kH;
#pragma unroll
      for (int e = 0; e < 8; ++e) a[e] = (_Float16)wr[8 * half + e];
    }
    A[t] = a;
  }

  // ---- Per-lane C-accumulator coefficients.
  // C/D layout: VGPR r, lane -> gate M = r + 8*half (within tile t), batch N = m.
  float cw[4][8], cb[4][8];
#pragma unroll
  for (int t = 0; t < 4; ++t) {
    const int g0 = 16 * t + 8 * half;
#pragma unroll
    for (int r = 0; r < 8; ++r) {
      cw[t][r] = w_ih[g0 + r];                    // I == 1
      cb[t][r] = b_ih[g0 + r] + b_hh[g0 + r];
    }
  }

  // ---- Recurrent state: lane holds hidden j = r + 8*half for batch m.
  float c_st[8], h_st[8];
#pragma unroll
  for (int r = 0; r < 8; ++r) { c_st[r] = 0.0f; h_st[r] = 0.0f; }

  // B operand: element e = h[hidden e + 8*half][batch m] for e<8, zero pad above.
  v16h bv = {};   // h0 == 0

  const float4* xp = (const float4*)(x + (size_t)batch * kT);
  float4 xbuf0 = xp[0];
  float4 xbuf1 = xp[1];

  for (int t0 = 0; t0 < kT; t0 += 8) {
    float4 xn0 = {}, xn1 = {};
    if (t0 + 8 < kT) {                 // double-buffered x prefetch
      xn0 = xp[(t0 >> 2) + 2];
      xn1 = xp[(t0 >> 2) + 3];
    }
    const float xs[8] = {xbuf0.x, xbuf0.y, xbuf0.z, xbuf0.w,
                         xbuf1.x, xbuf1.y, xbuf1.z, xbuf1.w};
#pragma unroll
    for (int s = 0; s < 8; ++s) {
      const float xv = xs[s];
      v8f C[4];
#pragma unroll
      for (int t = 0; t < 4; ++t) {
#pragma unroll
        for (int r = 0; r < 8; ++r)
          C[t][r] = __builtin_fmaf(xv, cw[t][r], cb[t][r]);
      }

      C[0] = __builtin_amdgcn_wmma_f32_16x16x32_f16(false, A[0], false, bv,
                                                    (short)0, C[0], false, false);
      C[1] = __builtin_amdgcn_wmma_f32_16x16x32_f16(false, A[1], false, bv,
                                                    (short)0, C[1], false, false);
      C[2] = __builtin_amdgcn_wmma_f32_16x16x32_f16(false, A[2], false, bv,
                                                    (short)0, C[2], false, false);
      C[3] = __builtin_amdgcn_wmma_f32_16x16x32_f16(false, A[3], false, bv,
                                                    (short)0, C[3], false, false);

#pragma unroll
      for (int r = 0; r < 8; ++r) {
        const float ig = sigmoid_fast(C[0][r]);
        const float fg = sigmoid_fast(C[1][r]);
        const float gg = tanh_fast(C[2][r]);
        const float og = sigmoid_fast(C[3][r]);
        const float c  = __builtin_fmaf(fg, c_st[r], ig * gg);
        c_st[r] = c;
        h_st[r] = og * tanh_fast(c);
      }

      v16h nb = {};
#pragma unroll
      for (int r = 0; r < 8; ++r) nb[r] = (_Float16)h_st[r];
      bv = nb;
    }
    xbuf0 = xn0;
    xbuf1 = xn1;
  }

  // ---- Final projection: out = h_last @ w_out.T + b_out
#pragma unroll
  for (int r = 0; r < 8; ++r)
    h_sm[waveInBlk * 16 + m][r + 8 * half] = h_st[r];
  __syncthreads();

  for (int idx = tid; idx < kBatchPerBlock * kO; idx += 256) {
    const int bl = idx / kO;
    const int o  = idx % kO;
    float acc = b_out[o];
#pragma unroll
    for (int j = 0; j < kH; ++j)
      acc = __builtin_fmaf(h_sm[bl][j], w_out[o * kH + j], acc);
    out[(size_t)(blockIdx.x * kBatchPerBlock + bl) * kO + o] = acc;
  }
}

extern "C" void kernel_launch(void* const* d_in, const int* in_sizes, int n_in,
                              void* d_out, int out_size, void* d_ws, size_t ws_size,
                              hipStream_t stream) {
  (void)in_sizes; (void)n_in; (void)out_size; (void)d_ws; (void)ws_size;
  const float* x     = (const float*)d_in[0];
  const float* w_ih  = (const float*)d_in[1];
  const float* w_hh  = (const float*)d_in[2];
  const float* b_ih  = (const float*)d_in[3];
  const float* b_hh  = (const float*)d_in[4];
  const float* w_out = (const float*)d_in[5];
  const float* b_out = (const float*)d_in[6];
  float* out = (float*)d_out;

  dim3 grid(kB / kBatchPerBlock);   // 16 blocks
  dim3 block(256);                  // 8 waves -> 128 batch per block
  lstm_recurrence_kernel<<<grid, block, 0, stream>>>(x, w_ih, w_hh, b_ih, b_hh,
                                                     w_out, b_out, out);
}
